// graph_encoder_9749575762649
// MI455X (gfx1250) — matrix operand order repaired and checked by reading, not compiled, not run on hardware
//
#include <hip/hip_runtime.h>

// ---------------------------------------------------------------------------
// Types for CDNA5 WMMA (wave32): v_wmma_f32_16x16x32_bf16
// ---------------------------------------------------------------------------
typedef __attribute__((ext_vector_type(16))) __bf16 v16bf;
typedef __attribute__((ext_vector_type(8)))  __bf16 v8bf;
typedef __attribute__((ext_vector_type(8)))  float  v8f;

static __device__ inline v16bf mk16(v8bf lo, v8bf hi) {
    union { v16bf v; v8bf h[2]; } u;
    u.h[0] = lo; u.h[1] = hi;
    return u.v;
}
static __device__ inline v8bf ld8(const unsigned short* p) { return *(const v8bf*)p; }

static __device__ inline unsigned short f32_to_bf16(float f) {
    unsigned int u = __float_as_uint(f);
    unsigned int r = (u + 0x7fffu + ((u >> 16) & 1u)) >> 16;
    return (unsigned short)r;
}

// CDNA5 async copy global->LDS (16B per lane), tracked by ASYNCcnt (GVS mode)
static __device__ inline void async_ld_b128(unsigned lds_addr,
                                            const unsigned short* sbase,
                                            unsigned voff_bytes) {
    asm volatile("global_load_async_to_lds_b128 %0, %1, %2"
                 :: "v"(lds_addr), "v"(voff_bytes), "s"(sbase)
                 : "memory");
}
static __device__ inline void wait_async0() {
    asm volatile("s_wait_asynccnt 0x0" ::: "memory");
}
static __device__ inline unsigned lds_addr_of(const unsigned short* p) {
    return (unsigned)(unsigned long long)p;   // low 32 bits = LDS byte offset
}

#define BATCH 16
#define NNODE 1024
#define HDIM  256
#define DHEAD 64
#define NENT  767
#define NREL  256
#define ROOTI 46
#define MROWS (BATCH * NNODE)   // 16384

// ---------------------------------------------------------------------------
// float [L][K][Nc] -> bf16 transposed [L][Nc][K]  (weights; done once)
// ---------------------------------------------------------------------------
__global__ void convt_kernel(const float* __restrict__ in,
                             unsigned short* __restrict__ out,
                             int K, int Nc, int total) {
    int i = blockIdx.x * blockDim.x + threadIdx.x;  // indexes out
    if (i >= total) return;
    int k = i % K;
    int rest = i / K;
    int n = rest % Nc;
    int l = rest / Nc;
    out[i] = f32_to_bf16(in[((size_t)l * K + k) * Nc + n]);
}

// ---------------------------------------------------------------------------
// Embedding gather: x[b,i,:] = ent/rel/root embeddings, f32 + bf16
// ---------------------------------------------------------------------------
__global__ __launch_bounds__(256) void embed_kernel(
    const int* __restrict__ ents, const int* __restrict__ rels,
    const float* __restrict__ ent_table, const float* __restrict__ rel_table,
    float* __restrict__ x, unsigned short* __restrict__ xb) {
    int node = blockIdx.x;
    int b = node >> 10, i = node & 1023;
    int tid = threadIdx.x;
    float v;
    if (i < NENT)              v = ent_table[(size_t)ents[b * NENT + i] * HDIM + tid];
    else if (i < NENT + NREL)  v = rel_table[(size_t)rels[b * NREL + (i - NENT)] * HDIM + tid];
    else                       v = rel_table[(size_t)ROOTI * HDIM + tid];
    size_t idx = (size_t)node * HDIM + tid;
    x[idx]  = v;
    xb[idx] = f32_to_bf16(v);
}

// ---------------------------------------------------------------------------
// bf16 WMMA GEMM: C[M,Nc] = A[M,K] @ B[K,Nc], B given TRANSPOSED as BT[Nc,K].
// Workgroup tile 128x64, 8 waves; double-buffered LDS filled by async copies.
// tmode=1: bf16 output written per-head transposed [B][4][64][1024] (for V).
// ---------------------------------------------------------------------------
#define ASTR 40
#define BSTR 40
__global__ __launch_bounds__(256) void gemm_bf16_kernel(
    const unsigned short* __restrict__ A,    // [M,K] bf16
    const unsigned short* __restrict__ BT,   // [Nc,K] bf16 (transposed)
    const float* __restrict__ bias,          // [Nc] or null
    const float* __restrict__ prelu,         // [Nc] or null
    float* __restrict__ Cf,                  // [M,Nc] f32 or null
    unsigned short* __restrict__ Cb,         // bf16 out or null
    int tmode, int Mrows, int Ncols, int K) {
    __shared__ unsigned short As[2][128 * ASTR];
    __shared__ unsigned short Bs[2][64 * BSTR];

    int tid = threadIdx.x;
    int w = tid >> 5, lane = tid & 31;
    int lrow = lane & 15, lhalf = lane >> 4;
    int m0 = blockIdx.y * 128;
    int n0 = blockIdx.x * 64;

    const unsigned short* Abase = A  + (size_t)m0 * K;
    const unsigned short* Bbase = BT + (size_t)n0 * K;

    // Per-thread staging coordinates (all transfers are contiguous 16B)
    int ar = tid >> 1, akof = (tid & 1) * 16;   // A tile 128x32, 16 elems/thread
    int bn = tid >> 2, bkof = (tid & 3) * 8;    // B tile 64x32,   8 elems/thread
    unsigned avoff = (unsigned)((ar * K + akof) * 2);
    unsigned bvoff = (unsigned)((bn * K + bkof) * 2);

    v8f acc[4] = {};
    int nk = K >> 5;

    // Prologue: stage tile 0
    {
        unsigned l0 = lds_addr_of(&As[0][ar * ASTR + akof]);
        async_ld_b128(l0,      Abase, avoff);
        async_ld_b128(l0 + 16, Abase, avoff + 16);
        async_ld_b128(lds_addr_of(&Bs[0][bn * BSTR + bkof]), Bbase, bvoff);
    }

    for (int it = 0; it < nk; ++it) {
        int cur = it & 1;
        wait_async0();
        __syncthreads();
        if (it + 1 < nk) {
            unsigned koff = (unsigned)((it + 1) << 6);   // 32 elems * 2B
            unsigned l0 = lds_addr_of(&As[cur ^ 1][ar * ASTR + akof]);
            async_ld_b128(l0,      Abase, avoff + koff);
            async_ld_b128(l0 + 16, Abase, avoff + koff + 16);
            async_ld_b128(lds_addr_of(&Bs[cur ^ 1][bn * BSTR + bkof]),
                          Bbase, bvoff + koff);
        }
        const unsigned short* as_ = As[cur];
        const unsigned short* bs_ = Bs[cur];
        int r = w * 16 + lrow;
        int k0a = lhalf * 8;
        v16bf af = mk16(ld8(&as_[r * ASTR + k0a]), ld8(&as_[r * ASTR + k0a + 16]));
        v16bf bf_[4];
#pragma unroll
        for (int nb = 0; nb < 4; ++nb) {
            int col = nb * 16 + lrow;
            int k0 = lhalf * 8;
            bf_[nb] = mk16(ld8(&bs_[col * BSTR + k0]), ld8(&bs_[col * BSTR + k0 + 16]));
        }
#pragma unroll
        for (int nb = 0; nb < 4; ++nb)
            acc[nb] = __builtin_amdgcn_wmma_f32_16x16x32_bf16(
                false, af, false, bf_[nb], (short)0, acc[nb], false, false);
    }

    // Epilogue: C layout -> (row = v + (lane>=16)*8, col = lane&15)
#pragma unroll
    for (int nb = 0; nb < 4; ++nb) {
        int col = n0 + nb * 16 + lrow;
        float bv = bias  ? bias[col]  : 0.0f;
        float al = prelu ? prelu[col] : 1.0f;
#pragma unroll
        for (int v = 0; v < 8; ++v) {
            int row = m0 + w * 16 + v + lhalf * 8;
            float val = acc[nb][v] + bv;
            if (prelu) val = (val >= 0.0f) ? val : al * val;
            if (Cf) Cf[(size_t)row * Ncols + col] = val;
            if (Cb) {
                size_t idx;
                if (tmode) {  // [B][4][64][1024]
                    int b = row >> 10, n = row & 1023;
                    idx = ((((size_t)b * 4 + (col >> 6)) * 64 + (col & 63)) << 10) + n;
                } else {
                    idx = (size_t)row * Ncols + col;
                }
                Cb[idx] = f32_to_bf16(val);
            }
        }
    }
}

// ---------------------------------------------------------------------------
// Flash attention: grid (N/128, HEADS, B); block 256 (8 waves, 16 rows/wave)
// V supplied transposed per head: Vt[B][4][64][1024]. Residual +x fused.
// ---------------------------------------------------------------------------
#define KSTR 72
#define VSTR 40
#define PSTR 40
__global__ __launch_bounds__(256) void attn_kernel(
    const unsigned short* __restrict__ Qb,
    const unsigned short* __restrict__ Kb,
    const unsigned short* __restrict__ Vt,
    const int* __restrict__ adjs,
    const float* __restrict__ xin,
    float* __restrict__ o) {
    __shared__ unsigned short Ks[2][32 * KSTR];
    __shared__ unsigned short Vs[2][64 * VSTR];
    __shared__ unsigned short Ps[8][16 * PSTR];

    int tid = threadIdx.x;
    int w = tid >> 5, lane = tid & 31;
    int lrow = lane & 15, lhalf = lane >> 4;
    int bq = blockIdx.z, hh = blockIdx.y;
    int iw = blockIdx.x * 128 + w * 16;

    // Q fragments (held in registers for the whole loop)
    v16bf aq[2];
    {
        const unsigned short* qp =
            Qb + ((size_t)(bq * NNODE + iw + lrow)) * HDIM + hh * DHEAD;
#pragma unroll
        for (int c = 0; c < 2; ++c) {
            int k0 = c * 32 + lhalf * 8;
            aq[c] = mk16(ld8(qp + k0), ld8(qp + k0 + 16));
        }
    }

    // Staging coordinates (contiguous 16B per lane)
    int kkey = tid >> 3, koff8 = (tid & 7) * 8;   // K tile 32x64
    int vd   = tid >> 2, vseg  = (tid & 3) * 8;   // V tile 64x32 (d-major)
    const unsigned short* Kbase = Kb + ((size_t)bq * NNODE) * HDIM + hh * DHEAD;
    const unsigned short* Vbase = Vt + (((size_t)bq * 4 + hh) << 16); // *64*1024
    unsigned kvoff = (unsigned)((kkey * HDIM + koff8) * 2);
    unsigned vvoff = (unsigned)(((vd << 10) + vseg) * 2);

    float mrow[8], lsum[8];
    v8f oacc[4] = {};
#pragma unroll
    for (int v = 0; v < 8; ++v) { mrow[v] = -3.0e38f; lsum[v] = 0.0f; }

    unsigned short* Pw = Ps[w];
    const int* adjrow = adjs + ((size_t)bq * NNODE) * NNODE;

    // Prologue: stage key-block 0
    async_ld_b128(lds_addr_of(&Ks[0][kkey * KSTR + koff8]), Kbase, kvoff);
    async_ld_b128(lds_addr_of(&Vs[0][vd * VSTR + vseg]),    Vbase, vvoff);

    for (int kb = 0; kb < NNODE / 32; ++kb) {
        int j0 = kb * 32;
        int cur = kb & 1;
        wait_async0();
        __syncthreads();
        if (kb + 1 < NNODE / 32) {
            async_ld_b128(lds_addr_of(&Ks[cur ^ 1][kkey * KSTR + koff8]),
                          Kbase + (size_t)(j0 + 32) * HDIM, kvoff);
            async_ld_b128(lds_addr_of(&Vs[cur ^ 1][vd * VSTR + vseg]),
                          Vbase + (j0 + 32), vvoff);
            // prefetch next adjacency block (biggest HBM stream)
            __builtin_prefetch(adjrow + (size_t)(iw + lrow) * NNODE +
                               (j0 + 32) + lhalf * 16, 0, 0);
        }
        const unsigned short* ks_ = Ks[cur];
        const unsigned short* vs_ = Vs[cur];

        // Scores for 32 keys = two 16x16 WMMA tiles (K = d = 64 -> 2 chunks)
        v8f sblk[2];
#pragma unroll
        for (int sb = 0; sb < 2; ++sb) {
            v8f s = {};
#pragma unroll
            for (int c = 0; c < 2; ++c) {
                int key = sb * 16 + lrow;
                int k0 = c * 32 + lhalf * 8;
                v16bf bk = mk16(ld8(&ks_[key * KSTR + k0]),
                                ld8(&ks_[key * KSTR + k0 + 16]));
                s = __builtin_amdgcn_wmma_f32_16x16x32_bf16(
                    false, aq[c], false, bk, (short)0, s, false, false);
            }
#pragma unroll
            for (int v = 0; v < 8; ++v) {
                int ri = iw + v + lhalf * 8;
                int cj = j0 + sb * 16 + lrow;
                float sv = s[v] * 0.0625f;   // 1/sqrt(256)
                if (adjrow[(size_t)ri * NNODE + cj] == 0) sv = -3.0e38f;
                s[v] = sv;
            }
            sblk[sb] = s;
        }

        // Online softmax over the 32 columns (16-lane shuffle reductions)
#pragma unroll
        for (int v = 0; v < 8; ++v) {
            float bm = fmaxf(sblk[0][v], sblk[1][v]);
#pragma unroll
            for (int d_ = 1; d_ < 16; d_ <<= 1) bm = fmaxf(bm, __shfl_xor(bm, d_, 32));
            float mnew = fmaxf(mrow[v], bm);
            float corr = __expf(mrow[v] - mnew);
            float p0 = __expf(sblk[0][v] - mnew);
            float p1 = __expf(sblk[1][v] - mnew);
            float psum = p0 + p1;
#pragma unroll
            for (int d_ = 1; d_ < 16; d_ <<= 1) psum += __shfl_xor(psum, d_, 32);
            lsum[v] = lsum[v] * corr + psum;
            mrow[v] = mnew;
            int prow = v + lhalf * 8;
            Pw[prow * PSTR + lrow]      = f32_to_bf16(p0);
            Pw[prow * PSTR + 16 + lrow] = f32_to_bf16(p1);
#pragma unroll
            for (int nb = 0; nb < 4; ++nb) oacc[nb][v] *= corr;
        }
        // Within-wave LDS ordering is in program order; just stop reordering.
        asm volatile("" ::: "memory");

        // P (A-frag 16x32) @ V (B-frags 32x64)
        {
            int k0 = lhalf * 8;
            v16bf pf = mk16(ld8(&Pw[lrow * PSTR + k0]),
                            ld8(&Pw[lrow * PSTR + k0 + 16]));
            v16bf vf[4];
#pragma unroll
            for (int nb = 0; nb < 4; ++nb) {
                int col = nb * 16 + lrow;
                int kv = lhalf * 8;
                vf[nb] = mk16(ld8(&vs_[col * VSTR + kv]),
                              ld8(&vs_[col * VSTR + kv + 16]));
            }
#pragma unroll
            for (int nb = 0; nb < 4; ++nb)
                oacc[nb] = __builtin_amdgcn_wmma_f32_16x16x32_bf16(
                    false, pf, false, vf[nb], (short)0, oacc[nb], false, false);
        }
    }

    // Normalize + residual, write f32
#pragma unroll
    for (int v = 0; v < 8; ++v) {
        int ri = iw + v + lhalf * 8;
        float inv = 1.0f / fmaxf(lsum[v], 1e-30f);
        size_t base = ((size_t)(bq * NNODE + ri)) * HDIM + hh * DHEAD;
#pragma unroll
        for (int nb = 0; nb < 4; ++nb) {
            size_t idx = base + nb * 16 + lrow;
            o[idx] = oacc[nb][v] * inv + xin[idx];
        }
    }
}

// ---------------------------------------------------------------------------
// LayerNorm (optional residual): out = LN(in [+ res]) * g + b
// ---------------------------------------------------------------------------
__global__ __launch_bounds__(256) void ln_kernel(
    const float* __restrict__ in, const float* __restrict__ res,
    const float* __restrict__ g, const float* __restrict__ bta,
    float* __restrict__ outf, unsigned short* __restrict__ outb) {
    __shared__ float red[16];
    int row = blockIdx.x, tid = threadIdx.x;
    size_t idx = (size_t)row * HDIM + tid;
    float v = in[idx];
    if (res) v += res[idx];
    float s1 = v, s2 = v * v;
#pragma unroll
    for (int d_ = 1; d_ < 32; d_ <<= 1) {
        s1 += __shfl_xor(s1, d_, 32);
        s2 += __shfl_xor(s2, d_, 32);
    }
    if ((tid & 31) == 0) { red[tid >> 5] = s1; red[8 + (tid >> 5)] = s2; }
    __syncthreads();
    float t1 = 0.0f, t2 = 0.0f;
#pragma unroll
    for (int i = 0; i < 8; ++i) { t1 += red[i]; t2 += red[8 + i]; }
    float mean = t1 * (1.0f / HDIM);
    float var = t2 * (1.0f / HDIM) - mean * mean;
    float y = (v - mean) * rsqrtf(var + 1e-5f) * g[tid] + bta[tid];
    if (outf) outf[idx] = y;
    if (outb) outb[idx] = f32_to_bf16(y);
}

// ---------------------------------------------------------------------------
// Finalize: d_out = [ x (B*N*H f32) | emask = 1.0 (B*N) ]
// ---------------------------------------------------------------------------
__global__ void finalize_kernel(const float* __restrict__ x,
                                float* __restrict__ out, int total, int nx) {
    int i = blockIdx.x * blockDim.x + threadIdx.x;
    if (i < total) out[i] = (i < nx) ? x[i] : 1.0f;
}

// ---------------------------------------------------------------------------
// Host orchestration
// ---------------------------------------------------------------------------
extern "C" void kernel_launch(void* const* d_in, const int* in_sizes, int n_in,
                              void* d_out, int out_size, void* d_ws, size_t ws_size,
                              hipStream_t stream) {
    const int*   ents      = (const int*)  d_in[0];
    const int*   rels      = (const int*)  d_in[1];
    const int*   adjs      = (const int*)  d_in[2];
    const float* ent_table = (const float*)d_in[3];
    const float* rel_table = (const float*)d_in[4];
    const float* Wq        = (const float*)d_in[5];
    const float* Wk        = (const float*)d_in[6];
    const float* Wv        = (const float*)d_in[7];
    const float* l1w       = (const float*)d_in[8];
    const float* l1b       = (const float*)d_in[9];
    const float* l2w       = (const float*)d_in[10];
    const float* l2b       = (const float*)d_in[11];
    const float* ln_g      = (const float*)d_in[12];
    const float* ln_b      = (const float*)d_in[13];
    const float* alpha     = (const float*)d_in[14];

    const size_t MH = (size_t)MROWS * HDIM;
    char* p = (char*)d_ws;
    float* x   = (float*)p;          p += MH * 4;
    float* o   = (float*)p;          p += MH * 4;
    float* t   = (float*)p;          p += MH * 4;
    float* h2  = (float*)p;          p += MH * 4;
    unsigned short* xb  = (unsigned short*)p; p += MH * 2;
    unsigned short* tb  = (unsigned short*)p; p += MH * 2;
    unsigned short* Qb  = (unsigned short*)p; p += MH * 2;
    unsigned short* Kb  = (unsigned short*)p; p += MH * 2;
    unsigned short* Vtp = (unsigned short*)p; p += MH * 2;   // [B][4][64][1024]
    unsigned short* h1b = (unsigned short*)p; p += MH * 4 * 2;
    unsigned short* wq  = (unsigned short*)p; p += (size_t)2 * HDIM * HDIM * 2;
    unsigned short* wk  = (unsigned short*)p; p += (size_t)2 * HDIM * HDIM * 2;
    unsigned short* wv  = (unsigned short*)p; p += (size_t)2 * HDIM * HDIM * 2;
    unsigned short* w1  = (unsigned short*)p; p += (size_t)2 * HDIM * 4 * HDIM * 2;
    unsigned short* w2  = (unsigned short*)p; p += (size_t)2 * 4 * HDIM * HDIM * 2;

    // Convert + transpose weights to bf16 [L][Nc][K]
    const int nQ = 2 * HDIM * HDIM;
    const int nM = 2 * HDIM * 4 * HDIM;
    convt_kernel<<<(nQ + 255) / 256, 256, 0, stream>>>(Wq,  wq, HDIM, HDIM, nQ);
    convt_kernel<<<(nQ + 255) / 256, 256, 0, stream>>>(Wk,  wk, HDIM, HDIM, nQ);
    convt_kernel<<<(nQ + 255) / 256, 256, 0, stream>>>(Wv,  wv, HDIM, HDIM, nQ);
    convt_kernel<<<(nM + 255) / 256, 256, 0, stream>>>(l1w, w1, HDIM, 4 * HDIM, nM);
    convt_kernel<<<(nM + 255) / 256, 256, 0, stream>>>(l2w, w2, 4 * HDIM, HDIM, nM);

    embed_kernel<<<MROWS, 256, 0, stream>>>(ents, rels, ent_table, rel_table, x, xb);

    for (int j = 0; j < 2; ++j) {
        const unsigned short* wqj = wq + (size_t)j * HDIM * HDIM;
        const unsigned short* wkj = wk + (size_t)j * HDIM * HDIM;
        const unsigned short* wvj = wv + (size_t)j * HDIM * HDIM;
        const unsigned short* w1j = w1 + (size_t)j * HDIM * 4 * HDIM;
        const unsigned short* w2j = w2 + (size_t)j * 4 * HDIM * HDIM;

        dim3 gqkv(HDIM / 64, MROWS / 128);
        gemm_bf16_kernel<<<gqkv, 256, 0, stream>>>(xb, wqj, nullptr, nullptr,
            nullptr, Qb, 0, MROWS, HDIM, HDIM);
        gemm_bf16_kernel<<<gqkv, 256, 0, stream>>>(xb, wkj, nullptr, nullptr,
            nullptr, Kb, 0, MROWS, HDIM, HDIM);
        gemm_bf16_kernel<<<gqkv, 256, 0, stream>>>(xb, wvj, nullptr, nullptr,
            nullptr, Vtp, 1, MROWS, HDIM, HDIM);   // per-head transposed out

        attn_kernel<<<dim3(NNODE / 128, 4, BATCH), 256, 0, stream>>>(
            Qb, Kb, Vtp, adjs, x, o);

        ln_kernel<<<MROWS, 256, 0, stream>>>(o, nullptr, ln_g + j * HDIM,
                                             ln_b + j * HDIM, t, tb);

        gemm_bf16_kernel<<<dim3(4 * HDIM / 64, MROWS / 128), 256, 0, stream>>>(
            tb, w1j, l1b + (size_t)j * 4 * HDIM, alpha + (size_t)j * 4 * HDIM,
            nullptr, h1b, 0, MROWS, 4 * HDIM, HDIM);

        gemm_bf16_kernel<<<dim3(HDIM / 64, MROWS / 128), 256, 0, stream>>>(
            h1b, w2j, l2b + (size_t)j * HDIM, nullptr,
            h2, nullptr, 0, MROWS, HDIM, 4 * HDIM);

        ln_kernel<<<MROWS, 256, 0, stream>>>(h2, t, ln_g + j * HDIM,
                                             ln_b + j * HDIM, x, xb);
    }

    finalize_kernel<<<(out_size + 255) / 256, 256, 0, stream>>>(
        x, (float*)d_out, out_size, (int)MH);
}
